// Prcl_Loss_75514114998572
// MI455X (gfx1250) — matrix operand order repaired and checked by reading, not compiled, not run on hardware
//
#include <hip/hip_runtime.h>
#include <hip/hip_bf16.h>
#include <math.h>

typedef __attribute__((ext_vector_type(16))) _Float16 v16h;
typedef __attribute__((ext_vector_type(8)))  float    v8f;

#define B_   16
#define C_   256
#define H_   64
#define W_   64
#define HW_  4096
#define P_   65536
#define S_   21
#define Q_   256
#define N_   256
#define TEMP_ 0.5f
#define THR_  0.97f
#define EPS_  1e-12f
#define LN2_  0.6931471805599453f

// ---- workspace layout (bytes) ----
static constexpr size_t OFF_MU_NORM = 0;                                  // P*C f32
static constexpr size_t OFF_VLIST   = OFF_MU_NORM + (size_t)P_ * C_ * 4;  // S*P i32
static constexpr size_t OFF_HLIST   = OFF_VLIST   + (size_t)S_ * P_ * 4;  // S*P i32
static constexpr size_t OFF_SEG     = OFF_HLIST   + (size_t)S_ * P_ * 4;  // P u8
static constexpr size_t OFF_VCOUNT  = OFF_SEG     + (size_t)P_;           // S i32 (pad 256)
static constexpr size_t OFF_HCOUNT  = OFF_VCOUNT  + 256;
static constexpr size_t OFF_SUMINV  = OFF_HCOUNT  + 256;                  // S*C f32
static constexpr size_t OFF_WSUM    = OFF_SUMINV  + (size_t)S_ * C_ * 4;
static constexpr size_t OFF_PMU     = OFF_WSUM    + (size_t)S_ * C_ * 4;  // normalized proto mu
static constexpr size_t OFF_PSIG    = OFF_PMU     + (size_t)S_ * C_ * 4;
static constexpr size_t OFF_CDF     = OFF_PSIG    + (size_t)S_ * C_ * 4;  // S*20 f32
static constexpr size_t OFF_TOTAL   = OFF_CDF     + (size_t)S_ * 20 * 4;
static constexpr size_t OFF_END     = OFF_TOTAL   + 256;
static constexpr size_t ZERO_WORDS  = (OFF_END - OFF_VCOUNT) / 4;

// ---- deterministic hash RNG ----
__device__ __forceinline__ unsigned hash3(unsigned a, unsigned b, unsigned c) {
    unsigned x = a * 0x9E3779B9u + b * 0x85EBCA6Bu + c * 0xC2B2AE35u + 0x27D4EB2Fu;
    x ^= x >> 16; x *= 0x7FEB352Du; x ^= x >> 15; x *= 0x846CA68Bu; x ^= x >> 16;
    return x;
}
__device__ __forceinline__ float rng01(unsigned a, unsigned b, unsigned c) {
    return (float)(hash3(a, b, c) >> 8) * (1.0f / 16777216.0f);
}

// sigma_f[p][c] gathered from original [B,C,H,W] layout (L2-resident: 64MB)
__device__ __forceinline__ float sigma_at(const float* __restrict__ sigma, int p, int c) {
    int b = p >> 12, hw = p & 4095;
    return sigma[((size_t)b * C_ + c) * (size_t)HW_ + hw];
}
// natural log via v_log_f32 (args here are in [~0.2, 2]: safe range)
__device__ __forceinline__ float fast_log(float x) {
    return __builtin_amdgcn_logf(x) * LN2_;
}

// ---------------------------------------------------------------------------
__global__ void k_zero(float* base, size_t nwords) {
    size_t i = (size_t)blockIdx.x * blockDim.x + threadIdx.x;
    if (i < nwords) base[i] = 0.0f;
}

// per-pixel L2-normalized mu, stored transposed [P,C] for contiguous gathers
__global__ void k_mu_norm(const float* __restrict__ mu, float* __restrict__ mu_norm) {
    __shared__ float red[256];
    int p = blockIdx.x, t = threadIdx.x;
    int b = p >> 12, hw = p & 4095;
    float x = mu[((size_t)b * C_ + t) * (size_t)HW_ + hw];
    red[t] = x * x;
    __syncthreads();
    for (int s = 128; s; s >>= 1) { if (t < s) red[t] += red[t + s]; __syncthreads(); }
    float inv = 1.0f / fmaxf(sqrtf(red[0]), EPS_);
    mu_norm[(size_t)p * C_ + t] = x * inv;
}

// per-segment valid/hard pixel lists + packed per-pixel segment id (0xFF = none)
__global__ void k_lists(const float* __restrict__ label, const float* __restrict__ mask,
                        const float* __restrict__ prob, unsigned char* __restrict__ seg,
                        int* __restrict__ vcount, int* __restrict__ hcount,
                        int* __restrict__ vlist,  int* __restrict__ hlist) {
    int p = blockIdx.x * blockDim.x + threadIdx.x;
    if (p >= P_) return;
    int b = p >> 12, hw = p & 4095;
    int sfound = -1;
    if (mask[(size_t)b * HW_ + hw] > 0.0f) {
        for (int s = 0; s < S_; ++s) {
            if (label[((size_t)b * S_ + s) * HW_ + hw] > 0.0f) { sfound = s; break; }
        }
    }
    seg[p] = (unsigned char)(sfound >= 0 ? sfound : 0xFF);
    if (sfound >= 0) {
        int vi = atomicAdd(&vcount[sfound], 1);
        vlist[(size_t)sfound * P_ + vi] = p;
        if (prob[((size_t)b * S_ + sfound) * HW_ + hw] < THR_) {
            int hi = atomicAdd(&hcount[sfound], 1);
            hlist[(size_t)sfound * P_ + hi] = p;
        }
    }
}

// prototype GEMMs on WMMA:  suminv[S,C] = wm @ (1/sig),  wsum[S,C] = wm @ (mu/sig)
// grid (C/16 col tiles, 2 row tiles, 16 K chunks), block = 1 wave (32)
__global__ void k_proto_gemm(const float* __restrict__ mu, const float* __restrict__ sigma,
                             const unsigned char* __restrict__ seg,
                             float* __restrict__ suminv, float* __restrict__ wsum) {
    int lane = threadIdx.x;
    int colBase = blockIdx.x * 16;
    int rowBase = blockIdx.y * 16;
    int p0base  = blockIdx.z * (P_ / 16);
    int kbA = (lane < 16) ? 0 : 16;    // contiguous k order: lanes 16-31 hold K 16..31
    int kbB = (lane < 16) ? 0 : 16;    // contiguous k order: lanes 16-31 hold K 16..31
    unsigned srow = (unsigned)(rowBase + (lane & 15));   // rows >= 21 never match a seg id
    int c = colBase + (lane & 15);
    const _Float16 h1 = (_Float16)1.0f, h0 = (_Float16)0.0f;
    v8f accS = {}, accW = {};
    for (int kk = 0; kk < P_ / 16; kk += 32) {
        int p0 = p0base + kk;
        v16h a, bS, bW;
        // A: segment mask tile from packed ids — two 8-byte loads, branch-free
        uint2 q0 = *(const uint2*)(seg + p0 + kbA);
        uint2 q1 = *(const uint2*)(seg + p0 + 8 + kbA);
#pragma unroll
        for (int e = 0; e < 4; ++e) {
            a[e]      = (((q0.x >> (8 * e)) & 0xFFu) == srow) ? h1 : h0;
            a[e + 4]  = (((q0.y >> (8 * e)) & 0xFFu) == srow) ? h1 : h0;
            a[e + 8]  = (((q1.x >> (8 * e)) & 0xFFu) == srow) ? h1 : h0;
            a[e + 12] = (((q1.y >> (8 * e)) & 0xFFu) == srow) ? h1 : h0;
        }
        // B: 16 consecutive pixels at fixed channel c -> 4x b128 loads per array
        int pB = p0 + kbB;                       // 16-aligned, stays within one batch
        int b = pB >> 12, hw = pB & 4095;
        const float4* sp = (const float4*)(sigma + ((size_t)b * C_ + c) * (size_t)HW_ + hw);
        const float4* mp = (const float4*)(mu    + ((size_t)b * C_ + c) * (size_t)HW_ + hw);
#pragma unroll
        for (int v = 0; v < 4; ++v) {
            float4 sv = sp[v];
            float4 mv = mp[v];
            float i0 = __builtin_amdgcn_rcpf(sv.x), i1 = __builtin_amdgcn_rcpf(sv.y);
            float i2 = __builtin_amdgcn_rcpf(sv.z), i3 = __builtin_amdgcn_rcpf(sv.w);
            bS[4 * v + 0] = (_Float16)i0;           bS[4 * v + 1] = (_Float16)i1;
            bS[4 * v + 2] = (_Float16)i2;           bS[4 * v + 3] = (_Float16)i3;
            bW[4 * v + 0] = (_Float16)(mv.x * i0);  bW[4 * v + 1] = (_Float16)(mv.y * i1);
            bW[4 * v + 2] = (_Float16)(mv.z * i2);  bW[4 * v + 3] = (_Float16)(mv.w * i3);
        }
        accS = __builtin_amdgcn_wmma_f32_16x16x32_f16(false, a, false, bS, (short)0, accS, false, false);
        accW = __builtin_amdgcn_wmma_f32_16x16x32_f16(false, a, false, bW, (short)0, accW, false, false);
    }
    int mhalf = (lane < 16) ? 0 : 8;     // f32 C/D layout
#pragma unroll
    for (int r = 0; r < 8; ++r) {
        int s = rowBase + r + mhalf;
        if (s < S_) {
            atomicAdd(&suminv[(size_t)s * C_ + c], accS[r]);
            atomicAdd(&wsum  [(size_t)s * C_ + c], accW[r]);
        }
    }
}

// proto_sigma = 1/suminv ; proto_mu = proto_sigma * wsum, then L2-normalize proto_mu
__global__ void k_proto_final(const float* __restrict__ suminv, const float* __restrict__ wsum,
                              float* __restrict__ pmu, float* __restrict__ psig) {
    __shared__ float red[256];
    int s = blockIdx.x, t = threadIdx.x;
    float si = suminv[(size_t)s * C_ + t];
    float ps = 1.0f / si;
    float pm = ps * wsum[(size_t)s * C_ + t];
    red[t] = pm * pm;
    __syncthreads();
    for (int st = 128; st; st >>= 1) { if (t < st) red[t] += red[t + st]; __syncthreads(); }
    float inv = 1.0f / fmaxf(sqrtf(red[0]), EPS_);
    pmu [(size_t)s * C_ + t] = pm * inv;
    psig[(size_t)s * C_ + t] = ps;
}

// proto-vs-proto MLS similarity -> softmax CDF for negative-class sampling
__global__ void k_proto_cdf(const float* __restrict__ pmu, const float* __restrict__ psig,
                            float* __restrict__ cdf) {
    __shared__ float red[256];
    __shared__ float sim[S_ - 1];
    int i = blockIdx.x, t = threadIdx.x;
    float pmi = pmu[(size_t)i * C_ + t];
    float psi = psig[(size_t)i * C_ + t];
    for (int o = 0; o < S_ - 1; ++o) {
        int j = (i + 1 + o) % S_;
        float d  = pmi - pmu[(size_t)j * C_ + t];
        float dn = psi + psig[(size_t)j * C_ + t];
        red[t] = d * d * __builtin_amdgcn_rcpf(dn) + fast_log(dn);
        __syncthreads();
        for (int st = 128; st; st >>= 1) { if (t < st) red[t] += red[t + st]; __syncthreads(); }
        if (t == 0) sim[o] = -0.5f * red[0] / (float)C_;
        __syncthreads();
    }
    if (t == 0) {
        float mx = -1e30f;
        for (int o = 0; o < S_ - 1; ++o) mx = fmaxf(mx, sim[o] / TEMP_);
        float w[S_ - 1], tot = 0.0f;
        for (int o = 0; o < S_ - 1; ++o) { w[o] = expf(sim[o] / TEMP_ - mx); tot += w[o]; }
        float cum = 0.0f;
        for (int o = 0; o < S_ - 1; ++o) { cum += w[o]; cdf[i * (S_ - 1) + o] = cum / tot; }
    }
}

// main contrastive kernel: one block per (segment i, query q); 8 waves cover 257 columns
__global__ void k_logits(const float* __restrict__ mu_norm, const float* __restrict__ sigma,
                         const float* __restrict__ pmu, const float* __restrict__ psig,
                         const float* __restrict__ cdf,
                         const int* __restrict__ vcount, const int* __restrict__ hcount,
                         const int* __restrict__ vlist,  const int* __restrict__ hlist,
                         float* __restrict__ total) {
    int i = blockIdx.y, q = blockIdx.x;
    int hc = hcount[i];
    if (hc <= 0) return;                       // empty hard set contributes 0
    __shared__ float am[C_], asg[C_], lgt[N_ + 1], red[256];
    __shared__ float scdf[S_ - 1];
    __shared__ int   scnt[S_];
    int t = threadIdx.x;
    if (t < S_ - 1) scdf[t] = cdf[i * (S_ - 1) + t];
    if (t < S_)     scnt[t] = vcount[t];
    // sample hard anchor; stage its mu_norm row + gathered sigma row into LDS
    // via gfx1250 async-to-LDS (ASYNCcnt) instead of load+ds_store
    float u1 = rng01((unsigned)i, (unsigned)q, 1u);
    int jq = min((int)(u1 * (float)hc), hc - 1);
    int ap = hlist[(size_t)i * P_ + jq];
    {
        unsigned lds_am = (unsigned)(uintptr_t)(&am[t]);     // low 32 bits = LDS offset
        const float* g1 = mu_norm + (size_t)ap * C_ + t;
        asm volatile("global_load_async_to_lds_b32 %0, %1, off"
                     :: "v"(lds_am), "v"(g1) : "memory");
        unsigned lds_as = (unsigned)(uintptr_t)(&asg[t]);
        const float* g2 = sigma + ((size_t)(ap >> 12) * C_ + t) * (size_t)HW_ + (ap & 4095);
        asm volatile("global_load_async_to_lds_b32 %0, %1, off"
                     :: "v"(lds_as), "v"(g2) : "memory");
        asm volatile("s_wait_asynccnt 0" ::: "memory");
    }
    __syncthreads();

    int wave = t >> 5, lane = t & 31;
    for (int j = wave; j < N_ + 1; j += 8) {
        float acc = 0.0f;
        if (j == 0) {                          // positive = prototype
            #pragma unroll
            for (int k = 0; k < 8; ++k) {
                int c = lane + 32 * k;
                float d  = am[c]  - pmu [(size_t)i * C_ + c];
                float dn = asg[c] + psig[(size_t)i * C_ + c];
                acc += d * d * __builtin_amdgcn_rcpf(dn) + fast_log(dn);
            }
        } else {                               // sampled negative pixel
            int n = j - 1;
            float u2 = rng01((unsigned)(i * 1024 + q), (unsigned)n, 2u);
            int o = 0;
            while (o < S_ - 2 && u2 > scdf[o]) ++o;
            int neg = (i + 1 + o) % S_;
            int cnt = scnt[neg];
            float u3 = rng01((unsigned)(i * 1024 + q), (unsigned)n, 3u);
            int jp = (cnt > 0) ? min((int)(u3 * (float)cnt), cnt - 1) : -1;
            int p  = (jp >= 0) ? vlist[(size_t)neg * P_ + jp] : 0;
            __builtin_prefetch(&mu_norm[(size_t)p * C_], 0, 0);   // global_prefetch_b8
            #pragma unroll
            for (int k = 0; k < 8; ++k) {
                int c = lane + 32 * k;
                float d  = am[c]  - mu_norm[(size_t)p * C_ + c];
                float dn = asg[c] + sigma_at(sigma, p, c);
                acc += d * d * __builtin_amdgcn_rcpf(dn) + fast_log(dn);
            }
        }
        for (int off = 16; off; off >>= 1) acc += __shfl_xor(acc, off, 32);
        if (lane == 0) lgt[j] = (-0.5f * acc / (float)C_) / TEMP_;
    }
    __syncthreads();
    // logsumexp over 257 logits
    float m = lgt[t];
    if (t == 0) m = fmaxf(m, lgt[N_]);
    red[t] = m;
    __syncthreads();
    for (int st = 128; st; st >>= 1) { if (t < st) red[t] = fmaxf(red[t], red[t + st]); __syncthreads(); }
    float M = red[0];
    __syncthreads();
    float e = expf(lgt[t] - M);
    if (t == 0) e += expf(lgt[N_] - M);
    red[t] = e;
    __syncthreads();
    for (int st = 128; st; st >>= 1) { if (t < st) red[t] += red[t + st]; __syncthreads(); }
    if (t == 0) {
        float lse = M + logf(red[0]);
        atomicAdd(total, (lse - lgt[0]) / (float)Q_);
    }
}

__global__ void k_finalize(const int* __restrict__ vcount, const float* __restrict__ total,
                           float* __restrict__ out) {
    if (threadIdx.x == 0 && blockIdx.x == 0) {
        int valid = 0;
        for (int s = 0; s < S_; ++s) valid += (vcount[s] > 0) ? 1 : 0;
        out[0] = total[0] / (float)max(valid, 1);
    }
}

// ---------------------------------------------------------------------------
extern "C" void kernel_launch(void* const* d_in, const int* in_sizes, int n_in,
                              void* d_out, int out_size, void* d_ws, size_t ws_size,
                              hipStream_t stream) {
    (void)in_sizes; (void)n_in; (void)out_size; (void)ws_size;
    const float* mu    = (const float*)d_in[0];
    const float* sigma = (const float*)d_in[1];
    const float* label = (const float*)d_in[2];
    const float* mask  = (const float*)d_in[3];
    const float* prob  = (const float*)d_in[4];
    char* ws = (char*)d_ws;

    float*         mu_norm = (float*)(ws + OFF_MU_NORM);
    int*           vlist   = (int*)  (ws + OFF_VLIST);
    int*           hlist   = (int*)  (ws + OFF_HLIST);
    unsigned char* seg     = (unsigned char*)(ws + OFF_SEG);
    int*           vcount  = (int*)  (ws + OFF_VCOUNT);
    int*           hcount  = (int*)  (ws + OFF_HCOUNT);
    float*         suminv  = (float*)(ws + OFF_SUMINV);
    float*         wsum    = (float*)(ws + OFF_WSUM);
    float*         pmu     = (float*)(ws + OFF_PMU);
    float*         psig    = (float*)(ws + OFF_PSIG);
    float*         cdf     = (float*)(ws + OFF_CDF);
    float*         total   = (float*)(ws + OFF_TOTAL);

    k_zero<<<(unsigned)((ZERO_WORDS + 255) / 256), 256, 0, stream>>>(
        (float*)(ws + OFF_VCOUNT), ZERO_WORDS);
    k_mu_norm<<<P_, 256, 0, stream>>>(mu, mu_norm);
    k_lists<<<P_ / 256, 256, 0, stream>>>(label, mask, prob, seg, vcount, hcount, vlist, hlist);
    k_proto_gemm<<<dim3(C_ / 16, 2, 16), 32, 0, stream>>>(mu, sigma, seg, suminv, wsum);
    k_proto_final<<<S_, 256, 0, stream>>>(suminv, wsum, pmu, psig);
    k_proto_cdf<<<S_, 256, 0, stream>>>(pmu, psig, cdf);
    k_logits<<<dim3(Q_, S_), 256, 0, stream>>>(mu_norm, sigma, pmu, psig, cdf,
                                               vcount, hcount, vlist, hlist, total);
    k_finalize<<<1, 32, 0, stream>>>(vcount, total, (float*)d_out);
}
